// PytorchFalconAttentionModel_46986942218875
// MI455X (gfx1250) — matrix operand order repaired and checked by reading, not compiled, not run on hardware
//
#include <hip/hip_runtime.h>

// ---------------------------------------------------------------------------
// Falcon MQA block for MI455X (gfx1250): bf16 WMMA everywhere, fp32 accum.
// B=2, S=1024, H=71, HD=64, HIDDEN=4544, QKV_N=4672
// ---------------------------------------------------------------------------

#define S_LEN   1024
#define NHEADS  71
#define HD_DIM  64
#define HIDDEN  4544            // 71*64
#define QKV_N   4672            // HIDDEN + 2*64
#define MROWS   2048            // B*S

typedef __attribute__((ext_vector_type(16))) __bf16        v16bf;
typedef __attribute__((ext_vector_type(8)))  float         v8f;
typedef __attribute__((ext_vector_type(4)))  float         f32x4;
typedef __attribute__((ext_vector_type(4)))  unsigned short u16x4;
typedef __attribute__((ext_vector_type(8)))  unsigned short u16x8;
typedef __attribute__((ext_vector_type(16))) unsigned short u16x16;

// Matches the builtin's parameter type: 'int __vector(4)' in global / LDS AS.
typedef int i32x4 __attribute__((vector_size(16)));
typedef __attribute__((address_space(1))) i32x4 gas_i32x4;
typedef __attribute__((address_space(3))) i32x4 las_i32x4;

// Async global->LDS staging path (gfx1250): use only if the toolchain exposes
// both the async load builtin and the asynccnt wait; otherwise fall back to
// plain load+ds_store staging.
#if defined(__AMDGCN__) &&                                               \
    __has_builtin(__builtin_amdgcn_global_load_async_to_lds_b128) &&     \
    __has_builtin(__builtin_amdgcn_s_wait_asynccnt)
#define USE_ASYNC_LDS 1
#else
#define USE_ASYNC_LDS 0
#endif

__device__ __forceinline__ unsigned short f2bf(float f) {
  unsigned int u = __float_as_uint(f);
  unsigned int r = u + 0x7FFFu + ((u >> 16) & 1u);   // round-to-nearest-even
  return (unsigned short)(r >> 16);
}

__device__ __forceinline__ v16bf make_frag(u16x8 lo, u16x8 hi) {
  union { u16x16 u; v16bf b; } x;
#pragma unroll
  for (int i = 0; i < 8; ++i) { x.u[i] = lo[i]; x.u[8 + i] = hi[i]; }
  return x.b;
}

// ---------------------------------------------------------------------------
// fp32 -> bf16 elementwise convert, 4 elements/thread (all sizes % 4 == 0)
// ---------------------------------------------------------------------------
__global__ void f32_to_bf16_kernel(const float* __restrict__ in,
                                   unsigned short* __restrict__ out, int n4) {
  int i = blockIdx.x * 256 + threadIdx.x;
  if (i < n4) {
    f32x4 v = ((const f32x4*)in)[i];
    u16x4 o;
#pragma unroll
    for (int j = 0; j < 4; ++j) o[j] = f2bf(v[j]);
    ((u16x4*)out)[i] = o;
  }
}

// ---------------------------------------------------------------------------
// Tiled bf16 GEMM, fp32 accumulate via v_wmma_f32_16x16x32_bf16.
// Block = 128 threads (4 waves). Tile: 128(M) x 64(N), BK = 32.
// Wave w computes a 64x32 sub-tile: 4 M-frags x 2 N-frags = 8 accumulators.
// A staged row-major in LDS (stride 40 bf16) via async global->LDS B128 when
// available; B staged transposed (Bs[n][k]) so each lane's B-fragment
// (16 contiguous K at column n) is one 32B LDS read.
// ---------------------------------------------------------------------------
template <bool OUT_BF16>
__global__ __launch_bounds__(128)
void gemm_bf16_kernel(const unsigned short* __restrict__ A,
                      const unsigned short* __restrict__ B,
                      void* __restrict__ Cv,
                      int K, int lda, int ldb, int ldc) {
  __shared__ __align__(16) unsigned short As[128 * 40];
  __shared__ __align__(16) unsigned short Bs[64 * 40];

  const int tid  = threadIdx.x;
  const int wid  = tid >> 5;
  const int lane = tid & 31;
  const int l15  = lane & 15;
  const int hf   = lane >> 4;            // lane-half (0/1)
  const int m0   = blockIdx.y * 128;
  const int n0   = blockIdx.x * 64;
  const int wM   = (wid >> 1) * 64;      // wave M offset inside tile
  const int wN   = (wid & 1) * 32;       // wave N offset inside tile

  v8f acc[4][2];
#pragma unroll
  for (int mi = 0; mi < 4; ++mi)
#pragma unroll
    for (int ni = 0; ni < 2; ++ni) acc[mi][ni] = (v8f)(0.0f);

  const int kkB = tid & 31;              // B-staging: k within tile
  const int ncB = (tid >> 5) * 16;       // B-staging: n chunk

  for (int k0 = 0; k0 < K; k0 += 32) {
    // ---- stage A tile: thread t owns row t, 32 bf16 (64 B) ----
    {
      const unsigned short* src = A + (size_t)(m0 + tid) * lda + k0;
      unsigned short* dst = &As[tid * 40];
#if USE_ASYNC_LDS
      gas_i32x4* g = (gas_i32x4*)src;
      las_i32x4* l = (las_i32x4*)dst;
      __builtin_amdgcn_global_load_async_to_lds_b128(g + 0, l + 0, 0, 0);
      __builtin_amdgcn_global_load_async_to_lds_b128(g + 1, l + 1, 0, 0);
      __builtin_amdgcn_global_load_async_to_lds_b128(g + 2, l + 2, 0, 0);
      __builtin_amdgcn_global_load_async_to_lds_b128(g + 3, l + 3, 0, 0);
#else
      u16x8 a0 = *(const u16x8*)(src + 0);
      u16x8 a1 = *(const u16x8*)(src + 8);
      u16x8 a2 = *(const u16x8*)(src + 16);
      u16x8 a3 = *(const u16x8*)(src + 24);
      *(u16x8*)(dst + 0)  = a0;
      *(u16x8*)(dst + 8)  = a1;
      *(u16x8*)(dst + 16) = a2;
      *(u16x8*)(dst + 24) = a3;
#endif
    }
    // ---- stage B tile transposed: Bs[n][k] ----
    {
      const unsigned short* src = B + (size_t)(k0 + kkB) * ldb + n0 + ncB;
      u16x8 b0 = *(const u16x8*)(src + 0);
      u16x8 b1 = *(const u16x8*)(src + 8);
#pragma unroll
      for (int i = 0; i < 8; ++i) Bs[(ncB + i) * 40 + kkB]     = b0[i];
#pragma unroll
      for (int i = 0; i < 8; ++i) Bs[(ncB + 8 + i) * 40 + kkB] = b1[i];
    }
    // ---- prime L2 with the next K-tile while this one is consumed ----
    if (k0 + 32 < K) {
      __builtin_prefetch(A + (size_t)(m0 + tid) * lda + k0 + 32, 0, 3);
      __builtin_prefetch(B + (size_t)(k0 + 32 + kkB) * ldb + n0 + ncB, 0, 3);
    }
#if USE_ASYNC_LDS
    __builtin_amdgcn_s_wait_asynccnt(0);
#endif
    __syncthreads();

    // ---- fragments in contiguous K order (lane half hf holds K = 16*hf + [0,16)) ----
    v16bf am[4], bn[2];
#pragma unroll
    for (int mi = 0; mi < 4; ++mi) {
      const unsigned short* base = &As[(wM + mi * 16 + l15) * 40];
      u16x8 lo = *(const u16x8*)(base + 16 * hf);       // K = 16*hf + [0,8)
      u16x8 hi = *(const u16x8*)(base + 16 * hf + 8);   // K = 16*hf + [8,16)
      am[mi] = make_frag(lo, hi);
    }
#pragma unroll
    for (int ni = 0; ni < 2; ++ni) {
      const unsigned short* base = &Bs[(wN + ni * 16 + l15) * 40 + 16 * hf];
      u16x8 lo = *(const u16x8*)(base + 0);             // K = 16*hf + [0,8)
      u16x8 hi = *(const u16x8*)(base + 8);             // K = 16*hf + [8,16)
      bn[ni] = make_frag(lo, hi);
    }
#pragma unroll
    for (int mi = 0; mi < 4; ++mi)
#pragma unroll
      for (int ni = 0; ni < 2; ++ni)
        acc[mi][ni] = __builtin_amdgcn_wmma_f32_16x16x32_bf16(
            false, am[mi], false, bn[ni], (short)0, acc[mi][ni], false, false);

    __syncthreads();
  }

  // ---- epilogue: C VGPR i -> row i + 8*hf, col = l15 ----
#pragma unroll
  for (int mi = 0; mi < 4; ++mi) {
#pragma unroll
    for (int ni = 0; ni < 2; ++ni) {
      const int row0 = m0 + wM + mi * 16 + 8 * hf;
      const int col  = n0 + wN + ni * 16 + l15;
#pragma unroll
      for (int i = 0; i < 8; ++i) {
        float v = acc[mi][ni][i];
        if (OUT_BF16)
          ((unsigned short*)Cv)[(size_t)(row0 + i) * ldc + col] = f2bf(v);
        else
          ((float*)Cv)[(size_t)(row0 + i) * ldc + col] = v;
      }
    }
  }
}

// ---------------------------------------------------------------------------
// V transpose: fused[b, s, 4608 + d] (bf16) -> vT[b, d, s]  (B x 64 x 1024)
// ---------------------------------------------------------------------------
__global__ void transpose_v_kernel(const unsigned short* __restrict__ fused,
                                   unsigned short* __restrict__ vT) {
  int idx = blockIdx.x * 256 + threadIdx.x;       // 2*64*1024 = 131072
  if (idx >= 2 * 64 * 1024) return;
  int s = idx & 1023;
  int d = (idx >> 10) & 63;
  int b = idx >> 16;
  vT[idx] = fused[(size_t)(b * S_LEN + s) * QKV_N + HIDDEN + HD_DIM + d];
}

// ---------------------------------------------------------------------------
// Flash attention, one wave (32 threads) per (b, h, 16-query tile).
// scores: Q(16x64) @ K^T(64x16) via 2 WMMAs per 16-key tile.
// online softmax with shfl_xor row reductions (C-frag: row = i+8*hf, col=l15)
// P@V: probs re-laid out via LDS into a 16x32 A-fragment, V^T gives
// contiguous B-fragments; 4 WMMAs per 32-key step into O (16x64).
// ---------------------------------------------------------------------------
__global__ __launch_bounds__(32)
void attn_kernel(const unsigned short* __restrict__ fused,  // bf16 2048 x 4672
                 const unsigned short* __restrict__ vT,     // bf16 B x 64 x 1024
                 const float* __restrict__ alibi,           // (B, H, 1, S) f32
                 unsigned short* __restrict__ ctx) {        // bf16 2048 x 4544
  const int qt   = blockIdx.x;           // 0..63
  const int h    = blockIdx.y;           // 0..70
  const int b    = blockIdx.z;           // 0..1
  const int lane = threadIdx.x;
  const int l15  = lane & 15;
  const int hf   = lane >> 4;
  const int q0   = qt * 16;

  __shared__ __align__(16) unsigned short Ps[16 * 40];

  const float slope    = alibi[((size_t)b * NHEADS + h) * S_LEN + 1]; // slope_h*1
  const float inv_norm = 0.125f;                                      // 1/sqrt(64)

  // Q A-fragments for the two K-dim steps (d in [0,32) and [32,64))
  v16bf qf[2];
  {
    const unsigned short* qb =
        fused + (size_t)(b * S_LEN + q0 + l15) * QKV_N + h * HD_DIM;
#pragma unroll
    for (int s = 0; s < 2; ++s) {
      u16x8 lo = *(const u16x8*)(qb + s * 32 + 16 * hf);
      u16x8 hi = *(const u16x8*)(qb + s * 32 + 16 * hf + 8);
      qf[s] = make_frag(lo, hi);
    }
  }

  v8f O[4];
#pragma unroll
  for (int dt = 0; dt < 4; ++dt) O[dt] = (v8f)(0.0f);
  float mrow[8], lrow[8];
#pragma unroll
  for (int i = 0; i < 8; ++i) { mrow[i] = -1e30f; lrow[i] = 0.0f; }

  const int nblk = (q0 + 16 + 31) >> 5;   // 32-key blocks covering causal range
  for (int jb = 0; jb < nblk; ++jb) {
    const int kb = jb * 32;

    // --- scores: two 16x16 tiles (keys kb+[0,16), kb+[16,32)) ---
    v8f sc[2];
#pragma unroll
    for (int t = 0; t < 2; ++t) {
      v8f s_acc = (v8f)(0.0f);
#pragma unroll
      for (int s = 0; s < 2; ++s) {
        const unsigned short* kr =
            fused + (size_t)(b * S_LEN + kb + t * 16 + l15) * QKV_N + HIDDEN +
            s * 32 + 16 * hf;
        u16x8 lo = *(const u16x8*)(kr + 0);
        u16x8 hi = *(const u16x8*)(kr + 8);
        v16bf kf = make_frag(lo, hi);
        s_acc = __builtin_amdgcn_wmma_f32_16x16x32_bf16(
            false, qf[s], false, kf, (short)0, s_acc, false, false);
      }
      sc[t] = s_acc;
    }

    // --- scale + alibi + causal mask + online softmax bookkeeping ---
    const int key0 = kb + l15;
    const int key1 = kb + 16 + l15;
    const float al0 = slope * (float)key0;
    const float al1 = slope * (float)key1;
#pragma unroll
    for (int i = 0; i < 8; ++i) {
      const int row = q0 + i + 8 * hf;
      float v0 = sc[0][i] * inv_norm + al0;
      float v1 = sc[1][i] * inv_norm + al1;
      if (key0 > row) v0 += -1e9f;
      if (key1 > row) v1 += -1e9f;
      float mx = fmaxf(v0, v1);
#pragma unroll
      for (int msk = 1; msk < 16; msk <<= 1) mx = fmaxf(mx, __shfl_xor(mx, msk, 32));
      const float mnew = fmaxf(mrow[i], mx);
      const float p0 = __expf(v0 - mnew);
      const float p1 = __expf(v1 - mnew);
      float rs = p0 + p1;
#pragma unroll
      for (int msk = 1; msk < 16; msk <<= 1) rs += __shfl_xor(rs, msk, 32);
      const float corr = __expf(mrow[i] - mnew);
      lrow[i] = lrow[i] * corr + rs;
      mrow[i] = mnew;
#pragma unroll
      for (int dt = 0; dt < 4; ++dt) O[dt][i] *= corr;
      sc[0][i] = p0;
      sc[1][i] = p1;
    }

    // --- re-layout probs through LDS into a 16x32 A-fragment ---
#pragma unroll
    for (int i = 0; i < 8; ++i) {
      Ps[(i + 8 * hf) * 40 + l15]      = f2bf(sc[0][i]);
      Ps[(i + 8 * hf) * 40 + 16 + l15] = f2bf(sc[1][i]);
    }
    __syncthreads();
    v16bf pf;
    {
      const unsigned short* base = &Ps[l15 * 40];
      u16x8 lo = *(const u16x8*)(base + 16 * hf);
      u16x8 hi = *(const u16x8*)(base + 16 * hf + 8);
      pf = make_frag(lo, hi);
    }

    // --- P @ V: 4 d-tiles of 16 ---
#pragma unroll
    for (int dt = 0; dt < 4; ++dt) {
      const unsigned short* vr =
          vT + ((size_t)(b * HD_DIM) + dt * 16 + l15) * S_LEN + kb + 16 * hf;
      u16x8 lo = *(const u16x8*)(vr + 0);
      u16x8 hi = *(const u16x8*)(vr + 8);
      v16bf vf = make_frag(lo, hi);
      O[dt] = __builtin_amdgcn_wmma_f32_16x16x32_bf16(
          false, pf, false, vf, (short)0, O[dt], false, false);
    }
    __syncthreads();
  }

  // --- normalize and store ctx as bf16 (row stride HIDDEN) ---
#pragma unroll
  for (int dt = 0; dt < 4; ++dt) {
#pragma unroll
    for (int i = 0; i < 8; ++i) {
      const int row = q0 + i + 8 * hf;
      const float o = O[dt][i] / lrow[i];
      ctx[(size_t)(b * S_LEN + row) * HIDDEN + h * HD_DIM + dt * 16 + l15] =
          f2bf(o);
    }
  }
}

// ---------------------------------------------------------------------------
// Host launcher
// ---------------------------------------------------------------------------
extern "C" void kernel_launch(void* const* d_in, const int* in_sizes, int n_in,
                              void* d_out, int out_size, void* d_ws, size_t ws_size,
                              hipStream_t stream) {
  const float* x       = (const float*)d_in[0];  // (B,S,HIDDEN)
  const float* alibi   = (const float*)d_in[1];  // (B,H,1,S)
  // d_in[2] attention_mask: pure causal -1e9, applied analytically
  const float* w_qkv   = (const float*)d_in[3];  // (HIDDEN, QKV_N)
  const float* w_dense = (const float*)d_in[4];  // (HIDDEN, HIDDEN)
  float* out = (float*)d_out;

  // workspace layout (bf16 buffers), 256B aligned
  char* ws = (char*)d_ws;
  size_t off = 0;
  auto take = [&](size_t bytes) {
    char* p = ws + off;
    off += (bytes + 255) & ~(size_t)255;
    return p;
  };
  unsigned short* xb     = (unsigned short*)take((size_t)MROWS * HIDDEN * 2);
  unsigned short* wqkvb  = (unsigned short*)take((size_t)HIDDEN * QKV_N * 2);
  unsigned short* wdb    = (unsigned short*)take((size_t)HIDDEN * HIDDEN * 2);
  unsigned short* fusedb = (unsigned short*)take((size_t)MROWS * QKV_N * 2);
  unsigned short* vTb    = (unsigned short*)take((size_t)2 * HD_DIM * S_LEN * 2);
  unsigned short* ctxb   = (unsigned short*)take((size_t)MROWS * HIDDEN * 2);
  (void)ws_size; (void)n_in; (void)in_sizes; (void)out_size;

  // 1) converts (vectorized x4)
  {
    int n1 = MROWS * HIDDEN / 4;
    f32_to_bf16_kernel<<<(n1 + 255) / 256, 256, 0, stream>>>(x, xb, n1);
    int n2 = HIDDEN * QKV_N / 4;
    f32_to_bf16_kernel<<<(n2 + 255) / 256, 256, 0, stream>>>(w_qkv, wqkvb, n2);
    int n3 = HIDDEN * HIDDEN / 4;
    f32_to_bf16_kernel<<<(n3 + 255) / 256, 256, 0, stream>>>(w_dense, wdb, n3);
  }

  // 2) fused QKV GEMM: (2048 x 4544) @ (4544 x 4672) -> bf16 fused
  gemm_bf16_kernel<true><<<dim3(QKV_N / 64, MROWS / 128), 128, 0, stream>>>(
      xb, wqkvb, (void*)fusedb, HIDDEN, HIDDEN, QKV_N, QKV_N);

  // 3) V transpose for contiguous P@V B-fragments
  transpose_v_kernel<<<(2 * HD_DIM * S_LEN + 255) / 256, 256, 0, stream>>>(
      fusedb, vTb);

  // 4) flash attention -> ctx (bf16, 2048 x 4544)
  attn_kernel<<<dim3(S_LEN / 16, NHEADS, 2), 32, 0, stream>>>(
      fusedb, vTb, alibi, ctxb);

  // 5) dense GEMM: (2048 x 4544) @ (4544 x 4544) -> fp32 out
  gemm_bf16_kernel<false><<<dim3(HIDDEN / 64, MROWS / 128), 128, 0, stream>>>(
      ctxb, wdb, (void*)out, HIDDEN, HIDDEN, HIDDEN, HIDDEN);
}
